// VanillaRNN_8555574854050
// MI455X (gfx1250) — matrix operand order repaired and checked by reading, not compiled, not run on hardware
//
#include <hip/hip_runtime.h>

typedef __attribute__((ext_vector_type(16))) __bf16 v16bf;
typedef __attribute__((ext_vector_type(8)))  float  v8f;

#define BATCH   2048
#define TSEQ    512
#define HID     1024
#define NCLS    10
#define BM      16              // batch rows per workgroup (WMMA M)
#define KC      32              // WMMA K per bf16 instruction
#define NKC     (HID / KC)      // 32 k-chunks
#define NTILES  (HID / 16)      // 64 output column tiles
#define WAVES   8
#define TPW     (NTILES / WAVES) // 8 tiles per wave

__device__ __forceinline__ unsigned short f2bf(float f) {
    unsigned int u = __float_as_uint(f);
    u += 0x7fffu + ((u >> 16) & 1u);      // round-to-nearest-even
    return (unsigned short)(u >> 16);
}
__device__ __forceinline__ float bf2f(unsigned short h) {
    return __uint_as_float(((unsigned int)h) << 16);
}

// ---------------------------------------------------------------------------
// Prepass: convert W_hh (fp32 [K][N] row-major) to bf16, pre-swizzled into the
// wave32 B-fragment layout so the RNN kernel loads 32 contiguous bytes/lane.
// Fragment (kc, nt): lane l, slot j  ->  K = kc*32 + (l>=16 ? 8 : 0) + (j&7) + (j>=8 ? 16 : 0),
//                                        N = nt*16 + (l & 15).
// ---------------------------------------------------------------------------
__global__ void swizzle_whh_kernel(const float* __restrict__ whh,
                                   unsigned short* __restrict__ frag) {
    int idx  = blockIdx.x * blockDim.x + threadIdx.x;   // destination order
    int j    =  idx        & 15;
    int lane = (idx >> 4)  & 31;
    int nt   = (idx >> 9)  & 63;
    int kc   =  idx >> 15;
    int k = kc * KC + ((lane >> 4) << 3) + (j & 7) + ((j >> 3) << 4);
    int n = nt * 16 + (lane & 15);
    frag[idx] = f2bf(whh[k * HID + n]);
}

// ---------------------------------------------------------------------------
// Persistent RNN: each workgroup owns 16 batch rows for all 512 timesteps.
// h lives in LDS in bf16 A-fragment layout:
//   element (m, k): kc=k>>5, k0=k&31, ahi=(k0>>3)&1, j=(k0&7)+((k0&16)?8:0),
//   lane' = ahi*16 + m,  addr = ((kc*32 + lane')*16 + j)
// ---------------------------------------------------------------------------
__global__ __launch_bounds__(256) void rnn_persistent_kernel(
    const float* __restrict__ x,            // [B][T]
    const float* __restrict__ whx,          // [1][H]
    const unsigned short* __restrict__ whhf,// bf16 fragment-swizzled [H][H]
    const float* __restrict__ wph,          // [H][C]
    const float* __restrict__ bh,           // [H]
    const float* __restrict__ bp,           // [C]
    float* __restrict__ out) {              // [B][C]
    __shared__ __align__(32) unsigned short hbuf[NKC * 32 * 16]; // 32 KB
    __shared__ float whx_s[HID];
    __shared__ float bh_s[HID];
    __shared__ float xrow[BM];

    const int tid  = threadIdx.x;
    const int wave = tid >> 5;
    const int lane = tid & 31;
    const int b0   = blockIdx.x * BM;
    const int hi   = lane >> 4;      // C/D row-half selector
    const int ncol = lane & 15;      // C/D column within tile

    for (int i = tid; i < HID; i += 256) { whx_s[i] = whx[i]; bh_s[i] = bh[i]; }
    for (int i = tid; i < NKC * 32 * 16; i += 256) hbuf[i] = 0;  // h0 = 0
    __syncthreads();

    for (int t = 0; t < TSEQ; ++t) {
        if (tid < BM) xrow[tid] = x[(size_t)(b0 + tid) * TSEQ + t];
        __syncthreads();   // h(t) writes + xrow visible

        // snapshot x for this step (read before the write-phase barrier)
        float xm[8];
        #pragma unroll
        for (int r = 0; r < 8; ++r) xm[r] = xrow[r + hi * 8];

        v8f acc[TPW] = {};
        for (int kc = 0; kc < NKC; ++kc) {
            v16bf a = *reinterpret_cast<const v16bf*>(&hbuf[(kc * 32 + lane) * 16]);
            const unsigned short* bbase =
                whhf + ((size_t)(kc * NTILES + wave * TPW) * 32 + lane) * 16;
            if (kc + 1 < NKC)
                __builtin_prefetch(
                    whhf + ((size_t)((kc + 1) * NTILES + wave * TPW) * 32 + lane) * 16,
                    0, 0);
            #pragma unroll
            for (int i = 0; i < TPW; ++i) {
                v16bf b = *reinterpret_cast<const v16bf*>(bbase + (size_t)i * (32 * 16));
                acc[i] = __builtin_amdgcn_wmma_f32_16x16x32_bf16(
                    false, a, false, b, (short)0, acc[i], false, false);
            }
        }

        __syncthreads();   // all waves done reading h(t) before overwrite

        #pragma unroll
        for (int i = 0; i < TPW; ++i) {
            int n   = (wave * TPW + i) * 16 + ncol;     // global column == next-step k
            float wx = whx_s[n];
            float bb = bh_s[n];
            int kc2 = n >> 5;
            int k0  = n & 31;
            int ahi = (k0 >> 3) & 1;
            int j   = (k0 & 7) + ((k0 & 16) ? 8 : 0);
            #pragma unroll
            for (int r = 0; r < 8; ++r) {
                int m   = r + hi * 8;
                float v = acc[i][r] + xm[r] * wx + bb;
                v = tanhf(v);
                hbuf[((kc2 * 32) + (ahi * 16 + m)) * 16 + j] = f2bf(v);
            }
        }
    }
    __syncthreads();

    // Final projection: out[b0+m][c] = sum_k h_T[m][k] * W_ph[k][c] + b_p[c]
    for (int idx = tid; idx < BM * NCLS; idx += 256) {
        int m = idx / NCLS;
        int c = idx - m * NCLS;
        float s = bp[c];
        for (int k = 0; k < HID; ++k) {
            int kc2 = k >> 5;
            int k0  = k & 31;
            int ahi = (k0 >> 3) & 1;
            int j   = (k0 & 7) + ((k0 & 16) ? 8 : 0);
            float hv = bf2f(hbuf[((kc2 * 32) + (ahi * 16 + m)) * 16 + j]);
            s += hv * wph[k * NCLS + c];
        }
        out[(size_t)(b0 + m) * NCLS + c] = s;
    }
}

extern "C" void kernel_launch(void* const* d_in, const int* in_sizes, int n_in,
                              void* d_out, int out_size, void* d_ws, size_t ws_size,
                              hipStream_t stream) {
    const float* x   = (const float*)d_in[0];   // [2048][512]
    const float* whx = (const float*)d_in[1];   // [1][1024]
    const float* whh = (const float*)d_in[2];   // [1024][1024]
    const float* wph = (const float*)d_in[3];   // [1024][10]
    const float* bh  = (const float*)d_in[4];   // [1024]
    const float* bp  = (const float*)d_in[5];   // [10]
    float* out = (float*)d_out;                 // [2048][10]

    unsigned short* whhf = (unsigned short*)d_ws;   // 2 MB bf16 fragments

    swizzle_whh_kernel<<<(HID * HID) / 256, 256, 0, stream>>>(whh, whhf);
    rnn_persistent_kernel<<<BATCH / BM, 256, 0, stream>>>(
        x, whx, whhf, wph, bh, bp, out);
}
